// ESNGenerator_2491081032383
// MI455X (gfx1250) — matrix operand order repaired and checked by reading, not compiled, not run on hardware
//
#include <hip/hip_runtime.h>
#include <hip/hip_bf16.h>

typedef __bf16 bf16_t;
typedef bf16_t v16bf __attribute__((ext_vector_type(16)));
typedef float  v8f   __attribute__((ext_vector_type(8)));

#define H_  512
#define M_  64
#define D_  64
#define T_  1024
#define N_  256
#define NB  16              // batch rows per workgroup (WMMA M)
#define KT_A (H_ / 32)      // 16 K-tiles for the recurrence GEMM
#define JT_H (H_ / 16)      // 32 output column tiles over H
#define TILE_ELEMS 512      // one 32x16 bf16 B-tile = 512 elements (32B/lane)

#define ABF_ELEMS (JT_H * KT_A * TILE_ELEMS)   // 262144 (512 KB)
#define CBF_ELEMS (JT_H * 2 * TILE_ELEMS)      // 32768  (64 KB)
#define WBF_ELEMS (4 * KT_A * TILE_ELEMS)      // 32768  (64 KB)

#if defined(__has_builtin)
#if __has_builtin(__builtin_amdgcn_tanhf)
#define TANHF(x) __builtin_amdgcn_tanhf(x)    // hardware V_TANH_F32 (CDNA5 TRANS op)
#endif
#endif
#ifndef TANHF
#define TANHF(x) tanhf(x)
#endif

// ---------------------------------------------------------------------------
// Pack A (HxH), C (HxM), W (DxH) fp32 -> bf16 tiles in WMMA B-operand layout.
// B tile is KxN = 32x16. Lane l supplies 16 contiguous bf16 (32 bytes):
//   n = l & 15 ; k = 8*(l>>4) + (i&7) + 16*(i>>3)   (i = 0..15 within the lane's 32B chunk)
// ---------------------------------------------------------------------------
__global__ __launch_bounds__(256) void esn_pack(
    const float* __restrict__ A, const float* __restrict__ C,
    const float* __restrict__ W,
    bf16_t* __restrict__ Abf, bf16_t* __restrict__ Cbf, bf16_t* __restrict__ Wbf)
{
    int id = blockIdx.x * 256 + threadIdx.x;
    if (id < ABF_ELEMS) {
        int tile = id >> 9, pos = id & 511;
        int jt = tile >> 4, kt = tile & 15;          // jt: h-tile, kt: k-tile
        int lane = pos >> 4, i = pos & 15;
        int n = lane & 15, k = ((lane >> 4) << 3) | (i & 7) | ((i >> 3) << 4);
        // out[n,h] = sum_k X[n,k] * A[h,k]  ->  B[k][h] = A[h,k]
        Abf[id] = (bf16_t)A[(jt * 16 + n) * H_ + kt * 32 + k];
    } else if (id < ABF_ELEMS + CBF_ELEMS) {
        int id2 = id - ABF_ELEMS;
        int tile = id2 >> 9, pos = id2 & 511;
        int jt = tile >> 1, kt = tile & 1;           // jt: h-tile, kt: m-tile
        int lane = pos >> 4, i = pos & 15;
        int n = lane & 15, k = ((lane >> 4) << 3) | (i & 7) | ((i >> 3) << 4);
        Cbf[id2] = (bf16_t)C[(jt * 16 + n) * M_ + kt * 32 + k];
    } else if (id < ABF_ELEMS + CBF_ELEMS + WBF_ELEMS) {
        int id3 = id - ABF_ELEMS - CBF_ELEMS;
        int tile = id3 >> 9, pos = id3 & 511;
        int dt = tile >> 4, kt = tile & 15;          // dt: d-tile, kt: h-tile
        int lane = pos >> 4, i = pos & 15;
        int n = lane & 15, k = ((lane >> 4) << 3) | (i & 7) | ((i >> 3) << 4);
        Wbf[id3] = (bf16_t)W[(dt * 16 + n) * H_ + kt * 32 + k];
    }
}

// ---------------------------------------------------------------------------
// Fused ESN recurrence. grid = 16 blocks, block = 256 threads = 8 waves.
// Block b owns batch rows [b*16, b*16+16). X_t lives in LDS (double buffer,
// bf16, A-operand layout). Wave w owns H column tiles {4w..4w+3}.
//
// A-operand 16x32 bf16 layout (ISA §7.12.2): element (m, k) lives at
//   lane = m + 16*((k>>3)&1) ; half = (k&7) + 8*(k>>4)
// ---------------------------------------------------------------------------
__global__ __launch_bounds__(256) void esn_recur(
    const bf16_t* __restrict__ Abf, const bf16_t* __restrict__ Cbf,
    const bf16_t* __restrict__ Wbf,
    const float* __restrict__ xi, const float* __restrict__ eta,
    float* __restrict__ out)
{
    __shared__ __attribute__((aligned(32))) bf16_t Xs[2 * NB * H_];  // 32 KB

    const int tid    = threadIdx.x;
    const int w      = tid >> 5;        // wave 0..7 (uniform per wave)
    const int lane   = tid & 31;
    const int rowsel = lane >> 4;       // 0/1
    const int r      = lane & 15;       // A-operand row / D-matrix column
    const int nb     = blockIdx.x * NB; // first batch row of this block

    const v8f zero = {0.f, 0.f, 0.f, 0.f, 0.f, 0.f, 0.f, 0.f};

    // X_{-1} = 0
    for (int i = tid; i < NB * H_; i += 256) Xs[i] = (bf16_t)0.0f;
    __syncthreads();

    for (int t = 0; t < T_; ++t) {
        // Opaque zero, regenerated every iteration: defeats LICM so the
        // packed-weight tile loads stay inside the loop (stream from L2)
        // instead of being hoisted and spilled to scratch.
        int tz;
        asm volatile("s_mov_b32 %0, 0" : "=s"(tz));
        const bf16_t* __restrict__ Abt = Abf + tz;
        const bf16_t* __restrict__ Cbt = Cbf + tz;
        const bf16_t* __restrict__ Wbt = Wbf + tz;

        const int rb = (t & 1) * (NB * H_);         // read  buffer (X_{t-1})
        const int wb = ((t & 1) ^ 1) * (NB * H_);   // write buffer (X_t)

        v8f acc[4];
        #pragma unroll
        for (int j = 0; j < 4; ++j) acc[j] = zero;

        // ---- fused input projection: acc += xi_block @ C^T ----
        const size_t xibase = ((size_t)(nb + r) * T_ + t) * M_;
        #pragma unroll
        for (int kt = 0; kt < 2; ++kt) {
            v16bf a;
            const int c0 = kt * 32 + 8 * rowsel;    // two contiguous 8-float runs
            #pragma unroll
            for (int i = 0; i < 8; ++i) a[i]     = (bf16_t)xi[xibase + c0 + i];
            #pragma unroll
            for (int i = 0; i < 8; ++i) a[8 + i] = (bf16_t)xi[xibase + c0 + 16 + i];
            #pragma unroll
            for (int j = 0; j < 4; ++j) {
                const int jg = w * 4 + j;
                const v16bf b = reinterpret_cast<const v16bf*>(
                    Cbt + (((size_t)jg * 2 + kt) << 9))[lane];
                acc[j] = __builtin_amdgcn_wmma_f32_16x16x32_bf16(
                    false, a, false, b, (short)0, acc[j], false, false);
            }
        }

        // ---- recurrence: acc += X_{t-1} @ A^T (X from LDS, A tiles from L2) ----
        #pragma unroll 2
        for (int kt = 0; kt < KT_A; ++kt) {
            const v16bf a =
                reinterpret_cast<const v16bf*>(&Xs[rb + kt * 512])[lane];
            #pragma unroll
            for (int j = 0; j < 4; ++j) {
                const int jg = w * 4 + j;
                const v16bf b = reinterpret_cast<const v16bf*>(
                    Abt + (((size_t)jg * 16 + kt) << 9))[lane];
                acc[j] = __builtin_amdgcn_wmma_f32_16x16x32_bf16(
                    false, a, false, b, (short)0, acc[j], false, false);
            }
        }

        // ---- hardware tanh, write X_t into LDS in A-operand layout ----
        #pragma unroll
        for (int j = 0; j < 4; ++j) {
            const int h      = (w * 4 + j) * 16 + r;   // this lane's h column
            const int kt     = h >> 5;
            const int kl     = h & 31;
            const int lane_s = 16 * ((kl >> 3) & 1);
            const int ioff   = (kl & 7) + 8 * (kl >> 4);
            #pragma unroll
            for (int v = 0; v < 8; ++v) {
                const int m = v + 8 * rowsel;          // D layout: M = v + 8*(l>>4)
                Xs[wb + kt * 512 + (m + lane_s) * 16 + ioff] =
                    (bf16_t)TANHF(acc[j][v]);
            }
        }
        __syncthreads();

        // ---- fused readout: waves 0..3 compute Z tile d-tile = w ----
        if (w < 4) {
            v8f z = zero;
            #pragma unroll 4
            for (int kt = 0; kt < KT_A; ++kt) {
                const v16bf a =
                    reinterpret_cast<const v16bf*>(&Xs[wb + kt * 512])[lane];
                const v16bf b = reinterpret_cast<const v16bf*>(
                    Wbt + (((size_t)w * 16 + kt) << 9))[lane];
                z = __builtin_amdgcn_wmma_f32_16x16x32_bf16(
                    false, a, false, b, (short)0, z, false, false);
            }
            #pragma unroll
            for (int v = 0; v < 8; ++v) {
                const int m = v + 8 * rowsel;
                const size_t idx =
                    ((size_t)(nb + m) * T_ + t) * D_ + w * 16 + r;
                out[idx] = z[v] + eta[idx];
            }
        }
    }
}

extern "C" void kernel_launch(void* const* d_in, const int* in_sizes, int n_in,
                              void* d_out, int out_size, void* d_ws, size_t ws_size,
                              hipStream_t stream) {
    const float* A   = (const float*)d_in[0];   // (H,H)
    const float* C   = (const float*)d_in[1];   // (H,M)
    const float* W   = (const float*)d_in[2];   // (D,H)
    const float* xi  = (const float*)d_in[3];   // (N,T,M)
    const float* eta = (const float*)d_in[4];   // (N,T,D)
    float* out = (float*)d_out;                 // (N,T,D) fp32

    bf16_t* Abf = (bf16_t*)d_ws;                // 512 KB
    bf16_t* Cbf = Abf + ABF_ELEMS;              //  64 KB
    bf16_t* Wbf = Cbf + CBF_ELEMS;              //  64 KB

    const int total = ABF_ELEMS + CBF_ELEMS + WBF_ELEMS;
    esn_pack<<<(total + 255) / 256, 256, 0, stream>>>(A, C, W, Abf, Cbf, Wbf);
    esn_recur<<<N_ / NB, 256, 0, stream>>>(Abf, Cbf, Wbf, xi, eta, out);
}
